// AttentionComparisonModel_27951647162822
// MI455X (gfx1250) — matrix operand order repaired and checked by reading, not compiled, not run on hardware
//
#include <hip/hip_runtime.h>
#include <hip/hip_bf16.h>

typedef __attribute__((ext_vector_type(16))) _Float16 v16h;
typedef __attribute__((ext_vector_type(8)))  _Float16 v8h;
typedef __attribute__((ext_vector_type(8)))  float    v8f;

#define NB   16      // batches
#define SS   2048    // sequence length
#define EE   256     // embedding dim
#define ROWS 64      // rows per workgroup (4 row-tiles of 16)
#define COLS 32      // columns per LDS block (2 col-tiles of 16)
#define AS_STRIDE 264  // halfs per A row (padded: 528B -> 4-bank skew)
#define BS_STRIDE 272  // halfs per B row (padded: 544B -> 8-bank skew)

// ---------------------------------------------------------------------------
// CDNA5 async global->LDS helpers. The low 32 bits of a flat LDS-aperture
// pointer are the wave-relative LDS byte offset (ISA 10.2), which is exactly
// what the VDST operand of global_load_async_to_lds wants. GVS addressing:
// mem_addr = SGPR_U64 + VGPR_I32 + IOFFSET. Tracked by ASYNCcnt.
// ---------------------------------------------------------------------------
__device__ __forceinline__ void async_b128(void* lds_ptr, unsigned goff,
                                           const void* sbase) {
  unsigned lds = (unsigned)(uintptr_t)lds_ptr;
  asm volatile("global_load_async_to_lds_b128 %0, %1, %2"
               :: "v"(lds), "v"(goff), "s"(sbase) : "memory");
}
__device__ __forceinline__ void wait_async_le4() {
  asm volatile("s_wait_asynccnt 0x4" ::: "memory");
}
__device__ __forceinline__ void wait_async_0() {
  asm volatile("s_wait_asynccnt 0x0" ::: "memory");
}

// ---------------------------------------------------------------------------
// Kernel 0: f32 -> f16 convert, per-row squared norm (of the rounded values,
// so the d2 diagonal lands on ~0), zero the colsum accumulator.
// One wave per row. Grid = B*S/8, block = 256 (8 waves).
// ---------------------------------------------------------------------------
__global__ void k_prep(const float* __restrict__ x, _Float16* __restrict__ xh,
                       float* __restrict__ sq, float* __restrict__ colsum) {
  const int row  = blockIdx.x * 8 + (threadIdx.x >> 5);
  const int lane = threadIdx.x & 31;
  const float* xp = x + (size_t)row * EE + lane * 8;

  v8h hv;
  float acc = 0.f;
#pragma unroll
  for (int i = 0; i < 8; ++i) {
    float v = xp[i];
    _Float16 h = (_Float16)v;
    hv[i] = h;
    float vf = (float)h;
    acc = fmaf(vf, vf, acc);
  }
  *(v8h*)(xh + (size_t)row * EE + lane * 8) = hv;

  acc += __shfl_xor(acc, 1, 32);
  acc += __shfl_xor(acc, 2, 32);
  acc += __shfl_xor(acc, 4, 32);
  acc += __shfl_xor(acc, 8, 32);
  acc += __shfl_xor(acc, 16, 32);
  if (lane == 0) sq[row] = acc;
  if (lane == 1) colsum[row] = 0.f;
}

// ---------------------------------------------------------------------------
// Kernel 1: Gram-matrix block via WMMA f16 -> double-exp weights -> rowsum
// (pass 0) -> colsum += u/rowsum (pass 1). Column blocks are double-buffered
// in LDS and filled with global_load_async_to_lds_b128.
// Grid = (S/ROWS, B), block = 256 (8 waves; wave w -> row-tile w&3, col-tile w>>2).
// ---------------------------------------------------------------------------
__global__ __launch_bounds__(256)
void k_attn(const _Float16* __restrict__ xh, const float* __restrict__ sq,
            const float* __restrict__ alpha_p, float* __restrict__ colsum) {
  __shared__ __align__(16) _Float16 As[ROWS * AS_STRIDE];        // 33792 B
  __shared__ __align__(16) _Float16 Bs[2][COLS * BS_STRIDE];     // 34816 B
  __shared__ float rowsum_s[ROWS];
  __shared__ float sqrow_s[ROWS];
  __shared__ float sqcol_s[COLS];
  __shared__ float colsum_s[COLS];

  const int b    = blockIdx.y;
  const int r0   = blockIdx.x * ROWS;
  const int tid  = threadIdx.x;
  const int lane = tid & 31;
  const int wave = tid >> 5;
  const int hlf  = lane >> 4;   // 0: lanes 0-15, 1: lanes 16-31
  const int l16  = lane & 15;
  const int rt   = wave & 3;    // row tile 0..3
  const int ct   = wave >> 2;   // col tile 0..1
  const float alpha = alpha_p[0];

  const _Float16* xb  = xh + (size_t)b * SS * EE;
  const float*    sqb = sq + (size_t)b * SS;

  // Async-stage the 64x256 row block (2048 16B chunks, 8 per thread).
#pragma unroll
  for (int i = 0; i < 8; ++i) {
    int c = tid + i * 256;
    int r = c >> 5;             // 32 chunks per 256-half row
    int o = (c & 31) * 8;
    async_b128(&As[r * AS_STRIDE + o],
               (unsigned)(((r0 + r) * EE + o) * 2), xb);
  }
  if (tid < ROWS) { sqrow_s[tid] = sqb[r0 + tid]; rowsum_s[tid] = 0.f; }

  // Per-lane A fragment base: 16-bit A 16x32 ISA layout.
  // lane<16: K {0..7, 16..23}; lane>=16: K {8..15, 24..31} (per 32-K step).
  const _Float16* arow = &As[(rt * 16 + l16) * AS_STRIDE + hlf * 8];

  float rsum[8];
#pragma unroll
  for (int i = 0; i < 8; ++i) rsum[i] = 0.f;

  for (int pass = 0; pass < 2; ++pass) {
    // Prologue: async-stage column block 0 into buffer 0.
#pragma unroll
    for (int i = 0; i < 4; ++i) {
      int c = tid + i * 256;
      int r = c >> 5;
      int o = (c & 31) * 8;
      async_b128(&Bs[0][r * BS_STRIDE + o], (unsigned)((r * EE + o) * 2), xb);
    }

    for (int it = 0; it < SS / COLS; ++it) {
      const int t0  = it * COLS;
      const int buf = it & 1;

      if (t0 + COLS < SS) {
        // Issue next block into the other buffer, then wait until only those
        // 4 per-wave async ops remain in flight (in-order completion =>
        // current block + everything older has landed in LDS).
#pragma unroll
        for (int i = 0; i < 4; ++i) {
          int c = tid + i * 256;
          int r = c >> 5;
          int o = (c & 31) * 8;
          async_b128(&Bs[buf ^ 1][r * BS_STRIDE + o],
                     (unsigned)(((t0 + COLS + r) * EE + o) * 2), xb);
        }
        wait_async_le4();
      } else {
        wait_async_0();
      }
      if (tid < COLS) { sqcol_s[tid] = sqb[t0 + tid]; colsum_s[tid] = 0.f; }
      __syncthreads();

      // B fragment base: 16-bit B 32x16 layout.
      // lane<16: col l16, K {0..7, 16..23}; lane>=16: col l16, K {8..15, 24..31} (per step).
      const _Float16* bcol = &Bs[buf][(ct * 16 + l16) * BS_STRIDE + hlf * 8];

      v8f acc = {};
#pragma unroll
      for (int ks = 0; ks < 8; ++ks) {
        v8h alo = *(const v8h*)(arow + ks * 32);
        v8h ahi = *(const v8h*)(arow + ks * 32 + 16);
        v8h blo = *(const v8h*)(bcol + ks * 32);
        v8h bhi = *(const v8h*)(bcol + ks * 32 + 16);
        v16h a = __builtin_shufflevector(alo, ahi, 0,1,2,3,4,5,6,7,8,9,10,11,12,13,14,15);
        v16h bm = __builtin_shufflevector(blo, bhi, 0,1,2,3,4,5,6,7,8,9,10,11,12,13,14,15);
        acc = __builtin_amdgcn_wmma_f32_16x16x32_f16(
            /*neg_a=*/false, a, /*neg_b=*/false, bm,
            /*c_mod=*/(short)0, acc, /*reuse_a=*/false, /*reuse_b=*/false);
      }

      // C layout: acc[i] = G(row rt*16 + hlf*8 + i, col ct*16 + l16).
      const float sqc = sqcol_s[ct * 16 + l16];
      if (pass == 0) {
#pragma unroll
        for (int i = 0; i < 8; ++i) {
          int rl = rt * 16 + hlf * 8 + i;
          float d2 = sqrow_s[rl] + sqc - 2.0f * acc[i];
          d2 = fmaxf(d2, 0.0f);
          float u = __expf(__expf(-alpha * d2));   // softmax over exp(-a*d2)
          rsum[i] += u;
        }
        __syncthreads();                           // all reads of Bs[buf] done
      } else {
        float csum = 0.f;
#pragma unroll
        for (int i = 0; i < 8; ++i) {
          int rl = rt * 16 + hlf * 8 + i;
          float d2 = sqrow_s[rl] + sqc - 2.0f * acc[i];
          d2 = fmaxf(d2, 0.0f);
          float u = __expf(__expf(-alpha * d2));
          csum = fmaf(u, rowsum_s[rl], csum);      // rowsum_s holds 1/rowsum here
        }
        csum += __shfl_xor(csum, 16, 32);          // fold rows 8..15 half onto 0..7
        if (hlf == 0) atomicAdd(&colsum_s[ct * 16 + l16], csum);
        __syncthreads();                           // also fences all Bs[buf] reads
        if (tid < COLS) atomicAdd(&colsum[(size_t)b * SS + t0 + tid], colsum_s[tid]);
        // no extra barrier: the flushing threads themselves re-zero colsum_s
        // next iteration before the post-load barrier.
      }
    }

    if (pass == 0) {
      // Reduce per-lane partial rowsums across the 16 lanes of each half.
#pragma unroll
      for (int i = 0; i < 8; ++i) {
        float r = rsum[i];
        r += __shfl_xor(r, 1, 32);
        r += __shfl_xor(r, 2, 32);
        r += __shfl_xor(r, 4, 32);
        r += __shfl_xor(r, 8, 32);
        if (l16 == 0) atomicAdd(&rowsum_s[rt * 16 + hlf * 8 + i], r);
      }
      __syncthreads();
      if (tid < ROWS) rowsum_s[tid] = 1.0f / rowsum_s[tid];  // reciprocal for pass 1
      __syncthreads();
    }
  }
}

// ---------------------------------------------------------------------------
// Kernel 2: pooled[b,e] = (1/S) * sum_t colsum[b,t] * x[b,t,e];
//           out[b,o]    = pooled[b,:] . W[o,:] + bias[o].
// Grid = B, block = 256.
// ---------------------------------------------------------------------------
__global__ void k_out(const float* __restrict__ x, const float* __restrict__ colsum,
                      const float* __restrict__ W, const float* __restrict__ bias,
                      float* __restrict__ out) {
  __shared__ float pooled[EE];
  const int b = blockIdx.x, tid = threadIdx.x;
  const float* xb = x + (size_t)b * SS * EE;
  const float* cs = colsum + (size_t)b * SS;

  float acc = 0.f;
  for (int t = 0; t < SS; ++t)
    acc = fmaf(cs[t], xb[(size_t)t * EE + tid], acc);
  pooled[tid] = acc * (1.0f / (float)SS);
  __syncthreads();

  float o = 0.f;
  const float* wr = W + (size_t)tid * EE;
#pragma unroll 4
  for (int e = 0; e < EE; ++e)
    o = fmaf(pooled[e], wr[e], o);
  out[(size_t)b * 256 + tid] = o + bias[tid];
}

// ---------------------------------------------------------------------------
extern "C" void kernel_launch(void* const* d_in, const int* in_sizes, int n_in,
                              void* d_out, int out_size, void* d_ws, size_t ws_size,
                              hipStream_t stream) {
  const float* x     = (const float*)d_in[0];  // (B,S,E) f32
  const float* alpha = (const float*)d_in[1];  // scalar
  const float* W     = (const float*)d_in[2];  // (OUT,E) f32
  const float* bias  = (const float*)d_in[3];  // (OUT,) f32
  float* out = (float*)d_out;

  char* ws = (char*)d_ws;
  _Float16* xh   = (_Float16*)ws;                                   // 16 MB
  float*    sq   = (float*)(ws + (size_t)NB * SS * EE * 2);         // 128 KB
  float*    csum = sq + (size_t)NB * SS;                            // 128 KB

  k_prep<<<dim3(NB * SS / 8), 256, 0, stream>>>(x, xh, sq, csum);
  k_attn<<<dim3(SS / ROWS, NB), 256, 0, stream>>>(xh, sq, alpha, csum);
  k_out<<<dim3(NB), 256, 0, stream>>>(x, csum, W, bias, out);
}
